// HeteroAPPNPEncoder_6760278524233
// MI455X (gfx1250) — matrix operand order repaired and checked by reading, not compiled, not run on hardware
//
#include <hip/hip_runtime.h>

// ---------------------------------------------------------------------------
// HeteroAPPNP encoder for MI455X (gfx1250, wave32).
//  - MLP layers via v_wmma_f32_16x16x32_bf16 (bf16 A/B, f32 accum)
//  - LayerNorm+ELU fused into GEMM epilogue (block covers full feature dim)
//  - PPR propagation as f32 atomic scatter-add (bandwidth bound: ~8GB total)
// ---------------------------------------------------------------------------

typedef __bf16 bf16_t;
typedef bf16_t v16bf __attribute__((ext_vector_type(16)));
typedef float  v8f   __attribute__((ext_vector_type(8)));

#define NUSER 50000
#define NITEM 50000
#define DIN   256
#define DH    512
#define DOUT  128
#define NEDGE 400000
#define KPPR  10
#define ALPHA 0.15f

// ---------------------------------------------------------------------------
// Weight pre-pack: W (K x N row-major f32) -> bf16 B-fragment order.
// B fragment (this code's contiguous k order, 32x16 KxN tile): lane<16 holds col n=lane, k=kt*32+0..15
//                                      lane>=16 holds col n=lane-16, k=kt*32+16..31
// Packed layout: [((kt*NT + nt)*32 + lane)*16 + e]
// ---------------------------------------------------------------------------
__global__ void pack_w_kernel(const float* __restrict__ W, bf16_t* __restrict__ P,
                              int K, int N) {
    int idx = blockIdx.x * blockDim.x + threadIdx.x;
    int total = (K / 32) * (N / 16) * 32;
    if (idx >= total) return;
    int lane = idx & 31;
    int tile = idx >> 5;
    int NT = N / 16;
    int nt = tile % NT;
    int kt = tile / NT;
    int n  = nt * 16 + (lane & 15);
    int k0 = kt * 32 + ((lane & 16) ? 16 : 0);
    bf16_t* dst = P + (size_t)idx * 16;
#pragma unroll
    for (int e = 0; e < 16; ++e)
        dst[e] = (bf16_t)W[(size_t)(k0 + e) * N + n];
}

// ---------------------------------------------------------------------------
// A-fragment loaders (contiguous k order used here, same as B pack):
//   lane<16 : row m=lane,    elems 0..15 -> k=kb..kb+15, kb=kt*32
//   lane>=16: row m=lane-16, same but kb=kt*32+16
// ---------------------------------------------------------------------------
__device__ inline v16bf load_a_frag(const float* __restrict__ A, int lda, int m, int kb) {
    union { v16bf v; bf16_t e[16]; } r;
    const float* p = A + (size_t)m * lda + kb;
    float t[16];
    *(float4*)(t + 0)  = *(const float4*)(p + 0);
    *(float4*)(t + 4)  = *(const float4*)(p + 4);
    *(float4*)(t + 8)  = *(const float4*)(p + 8);
    *(float4*)(t + 12) = *(const float4*)(p + 12);
#pragma unroll
    for (int e = 0; e < 16; ++e) r.e[e] = (bf16_t)t[e];
    return r.v;
}

__device__ inline v16bf load_a_frag(const bf16_t* __restrict__ A, int lda, int m, int kb) {
    union { v16bf v; uint4 q[2]; } r;
    const bf16_t* p = A + (size_t)m * lda + kb;
    r.q[0] = *(const uint4*)(p);
    r.q[1] = *(const uint4*)(p + 8);
    return r.v;
}

__device__ inline v16bf load_b_frag(const bf16_t* __restrict__ Bp, int NT, int kt,
                                    int nt, int lane) {
    union { v16bf v; uint4 q[2]; } r;
    const bf16_t* p = Bp + ((size_t)(kt * NT + nt) * 32 + lane) * 16;
    r.q[0] = *(const uint4*)(p);
    r.q[1] = *(const uint4*)(p + 8);
    return r.v;
}

// ---------------------------------------------------------------------------
// Fused GEMM (M=16 rows per block, N=512 full width) + bias + LayerNorm + ELU.
// 8 waves; wave w owns N-subtiles [4w..4w+3]. Output bf16 for next layer.
// ---------------------------------------------------------------------------
template <typename AT, int KDIM>
__global__ void __launch_bounds__(256)
gemm512_ln_elu(const AT* __restrict__ A, const bf16_t* __restrict__ Bp,
               const float* __restrict__ bias, const float* __restrict__ gamma,
               const float* __restrict__ beta, bf16_t* __restrict__ Out) {
    const int N = 512, NT = N / 16;
    int tid  = threadIdx.x;
    int wave = tid >> 5;
    int lane = tid & 31;
    int rowBase = blockIdx.x * 16;
    int m   = rowBase + (lane & 15);
    int nt0 = wave * 4;

    v8f zero = {0.f, 0.f, 0.f, 0.f, 0.f, 0.f, 0.f, 0.f};
    v8f acc[4] = {zero, zero, zero, zero};

    for (int kt = 0; kt < KDIM / 32; ++kt) {
        int kb = kt * 32 + ((lane & 16) ? 16 : 0);
        v16bf a = load_a_frag(A, KDIM, m, kb);
#pragma unroll
        for (int j = 0; j < 4; ++j) {
            v16bf b = load_b_frag(Bp, NT, kt, nt0 + j, lane);
            acc[j] = __builtin_amdgcn_wmma_f32_16x16x32_bf16(
                false, a, false, b, (short)0, acc[j], false, false);
        }
    }

    // Spill D + bias to LDS (16 rows x 512 cols, padded stride).
    __shared__ float tile[16 * 520];
    int mr = (lane & 16) ? 8 : 0;
#pragma unroll
    for (int j = 0; j < 4; ++j) {
        int n = (nt0 + j) * 16 + (lane & 15);
#pragma unroll
        for (int r = 0; r < 8; ++r)
            tile[(mr + r) * 520 + n] = acc[j][r] + bias[n];
    }
    __syncthreads();

    // LayerNorm + ELU: 16 threads per row (half-wave), 32 cols each.
    int row = tid >> 4;   // 0..15
    int c0  = tid & 15;
    float sum = 0.f, sq = 0.f;
#pragma unroll
    for (int jj = 0; jj < 32; ++jj) {
        float v = tile[row * 520 + c0 + jj * 16];
        sum += v; sq += v * v;
    }
#pragma unroll
    for (int off = 8; off >= 1; off >>= 1) {
        sum += __shfl_xor(sum, off, 16);
        sq  += __shfl_xor(sq,  off, 16);
    }
    float mean = sum * (1.0f / 512.0f);
    float var  = sq * (1.0f / 512.0f) - mean * mean;
    float rstd = rsqrtf(var + 1e-5f);

    bf16_t* orow = Out + (size_t)(rowBase + row) * 512;
#pragma unroll
    for (int jj = 0; jj < 32; ++jj) {
        int n = c0 + jj * 16;
        float v = (tile[row * 520 + n] - mean) * rstd * gamma[n] + beta[n];
        v = (v > 0.f) ? v : (expm1f(v));   // ELU(alpha=1)
        orow[n] = (bf16_t)v;
    }
}

// ---------------------------------------------------------------------------
// Final projection: H2 (N x 512 bf16) @ W3 -> f32 h0 (and initial h).
// 8 waves x 16 cols = 128 output cols; one 16x16 WMMA tile per wave per kstep.
// ---------------------------------------------------------------------------
__global__ void __launch_bounds__(256)
gemm128_bias(const bf16_t* __restrict__ A, const bf16_t* __restrict__ Bp,
             const float* __restrict__ bias, float* __restrict__ H0,
             float* __restrict__ Hinit) {
    const int KDIM = 512, NT = DOUT / 16;
    int tid  = threadIdx.x;
    int wave = tid >> 5;
    int lane = tid & 31;
    int rowBase = blockIdx.x * 16;
    int m  = rowBase + (lane & 15);
    int nt = wave;

    v8f acc = {0.f, 0.f, 0.f, 0.f, 0.f, 0.f, 0.f, 0.f};
    for (int kt = 0; kt < KDIM / 32; ++kt) {
        int kb = kt * 32 + ((lane & 16) ? 16 : 0);
        v16bf a = load_a_frag(A, KDIM, m, kb);
        v16bf b = load_b_frag(Bp, NT, kt, nt, lane);
        acc = __builtin_amdgcn_wmma_f32_16x16x32_bf16(
            false, a, false, b, (short)0, acc, false, false);
    }
    int n  = nt * 16 + (lane & 15);
    int mr = (lane & 16) ? 8 : 0;
    float bn = bias[n];
#pragma unroll
    for (int r = 0; r < 8; ++r) {
        float v = acc[r] + bn;
        size_t o = (size_t)(rowBase + mr + r) * DOUT + n;
        H0[o]    = v;
        Hinit[o] = v;
    }
}

// ---------------------------------------------------------------------------
// Propagation kernels (bandwidth / atomic bound).
// ---------------------------------------------------------------------------
__global__ void zero_f32(float* __restrict__ p, long long n) {
    long long i = (long long)blockIdx.x * blockDim.x + threadIdx.x;
    if (i < n) p[i] = 0.f;
}

__global__ void count_edges(const int* __restrict__ dst, float* __restrict__ cnt, int E) {
    int i = blockIdx.x * blockDim.x + threadIdx.x;
    if (i < E) atomicAdd(&cnt[dst[i]], 1.0f);
}

__global__ void scatter_add(const float* __restrict__ h, const int* __restrict__ src,
                            const int* __restrict__ dst, float* __restrict__ agg, int E) {
    long long i = (long long)blockIdx.x * blockDim.x + threadIdx.x;
    int e = (int)(i >> 7);
    int f = (int)(i & 127);
    if (e < E)
        atomicAdd(&agg[(size_t)dst[e] * DOUT + f], h[(size_t)src[e] * DOUT + f]);
}

__global__ void blend(const float* __restrict__ agg, const float* __restrict__ cnt,
                      const float* __restrict__ h0, float* __restrict__ h, long long n) {
    long long i = (long long)blockIdx.x * blockDim.x + threadIdx.x;
    if (i >= n) return;
    int node = (int)(i >> 7);
    float c = cnt[node];
    c = (c < 1.0f) ? 1.0f : c;
    h[i] = (1.0f - ALPHA) * agg[i] / c + ALPHA * h0[i];
}

// ---------------------------------------------------------------------------
// Host launcher.
// ---------------------------------------------------------------------------
static inline int cdiv_ll(long long a, long long b) { return (int)((a + b - 1) / b); }

extern "C" void kernel_launch(void* const* d_in, const int* in_sizes, int n_in,
                              void* d_out, int out_size, void* d_ws, size_t ws_size,
                              hipStream_t stream) {
    (void)in_sizes; (void)n_in; (void)out_size; (void)ws_size;

    const float* x_user = (const float*)d_in[0];
    const float* x_item = (const float*)d_in[1];
    const int*   ei_ui  = (const int*)d_in[2];   // [0:E)=user src, [E:2E)=item dst
    const int*   ei_iu  = (const int*)d_in[3];   // [0:E)=item src, [E:2E)=user dst

    const float* up[10];  // W1,b1,g1,be1,W2,b2,g2,be2,W3,b3
    const float* ip[10];
    for (int i = 0; i < 10; ++i) {
        up[i] = (const float*)d_in[4 + i];
        ip[i] = (const float*)d_in[14 + i];
    }

    // Workspace carve-up (256B aligned slabs).
    char* ws = (char*)d_ws;
    size_t off = 0;
    auto carve = [&](size_t bytes) -> char* {
        char* p = ws + off;
        off += (bytes + 255) & ~(size_t)255;
        return p;
    };
    bf16_t* Wu1 = (bf16_t*)carve((size_t)DIN * DH * 2);
    bf16_t* Wu2 = (bf16_t*)carve((size_t)DH * DH * 2);
    bf16_t* Wu3 = (bf16_t*)carve((size_t)DH * DOUT * 2);
    bf16_t* Wi1 = (bf16_t*)carve((size_t)DIN * DH * 2);
    bf16_t* Wi2 = (bf16_t*)carve((size_t)DH * DH * 2);
    bf16_t* Wi3 = (bf16_t*)carve((size_t)DH * DOUT * 2);
    bf16_t* H1  = (bf16_t*)carve((size_t)NUSER * DH * 2);   // reused user/item
    bf16_t* H2  = (bf16_t*)carve((size_t)NUSER * DH * 2);
    float*  h0u = (float*)carve((size_t)NUSER * DOUT * 4);
    float*  h0i = (float*)carve((size_t)NITEM * DOUT * 4);
    float*  agg_u = (float*)carve((size_t)NUSER * DOUT * 4);
    float*  agg_i = (float*)carve((size_t)NITEM * DOUT * 4);
    float*  cnt_u = (float*)carve((size_t)NUSER * 4);
    float*  cnt_i = (float*)carve((size_t)NITEM * 4);

    float* hu = (float*)d_out;                     // [NUSER, 128]
    float* hi = hu + (size_t)NUSER * DOUT;         // [NITEM, 128]

    // --- Pack weights to bf16 fragment order ---
    auto pack = [&](const float* W, bf16_t* P, int K, int N) {
        int total = (K / 32) * (N / 16) * 32;
        pack_w_kernel<<<cdiv_ll(total, 256), 256, 0, stream>>>(W, P, K, N);
    };
    pack(up[0], Wu1, DIN, DH);
    pack(up[4], Wu2, DH, DH);
    pack(up[8], Wu3, DH, DOUT);
    pack(ip[0], Wi1, DIN, DH);
    pack(ip[4], Wi2, DH, DH);
    pack(ip[8], Wi3, DH, DOUT);

    // --- MLPs (user then item, reusing H1/H2) ---
    const int gridM = NUSER / 16;  // 3125 (exact)
    gemm512_ln_elu<float, DIN><<<gridM, 256, 0, stream>>>(x_user, Wu1, up[1], up[2], up[3], H1);
    gemm512_ln_elu<bf16_t, DH><<<gridM, 256, 0, stream>>>(H1, Wu2, up[5], up[6], up[7], H2);
    gemm128_bias<<<gridM, 256, 0, stream>>>(H2, Wu3, up[9], h0u, hu);

    gemm512_ln_elu<float, DIN><<<gridM, 256, 0, stream>>>(x_item, Wi1, ip[1], ip[2], ip[3], H1);
    gemm512_ln_elu<bf16_t, DH><<<gridM, 256, 0, stream>>>(H1, Wi2, ip[5], ip[6], ip[7], H2);
    gemm128_bias<<<gridM, 256, 0, stream>>>(H2, Wi3, ip[9], h0i, hi);

    // --- Degree counts (edge-structure constants) ---
    zero_f32<<<cdiv_ll(NITEM, 256), 256, 0, stream>>>(cnt_i, NITEM);
    zero_f32<<<cdiv_ll(NUSER, 256), 256, 0, stream>>>(cnt_u, NUSER);
    count_edges<<<cdiv_ll(NEDGE, 256), 256, 0, stream>>>(ei_ui + NEDGE, cnt_i, NEDGE);
    count_edges<<<cdiv_ll(NEDGE, 256), 256, 0, stream>>>(ei_iu + NEDGE, cnt_u, NEDGE);

    // --- K PPR steps: both scatters read old h, then both blends write ---
    const long long nFu = (long long)NUSER * DOUT;
    const long long nFi = (long long)NITEM * DOUT;
    const long long nEF = (long long)NEDGE * DOUT;
    for (int k = 0; k < KPPR; ++k) {
        zero_f32<<<cdiv_ll(nFi, 256), 256, 0, stream>>>(agg_i, nFi);
        zero_f32<<<cdiv_ll(nFu, 256), 256, 0, stream>>>(agg_u, nFu);
        scatter_add<<<cdiv_ll(nEF, 256), 256, 0, stream>>>(hu, ei_ui, ei_ui + NEDGE, agg_i, NEDGE);
        scatter_add<<<cdiv_ll(nEF, 256), 256, 0, stream>>>(hi, ei_iu, ei_iu + NEDGE, agg_u, NEDGE);
        blend<<<cdiv_ll(nFi, 256), 256, 0, stream>>>(agg_i, cnt_i, h0i, hi, nFi);
        blend<<<cdiv_ll(nFu, 256), 256, 0, stream>>>(agg_u, cnt_u, h0u, hu, nFu);
    }
}
